// CausalSelfAttention_1778116461114
// MI455X (gfx1250) — matrix operand order repaired and checked by reading, not compiled, not run on hardware
//
#include <hip/hip_runtime.h>
#include <hip/hip_bf16.h>

#define DM 1024
#define NH 16
#define DH 64
#define BB 2
#define TT 2048
#define MM (BB*TT) /* 4096 */

typedef __bf16 bf16_t;
typedef __attribute__((ext_vector_type(16))) __bf16 v16bf;
typedef __attribute__((ext_vector_type(8)))  __bf16 v8bf;
typedef __attribute__((ext_vector_type(8)))  float  v8f;

// ---- fused DPP row_ror reductions within 16-lane groups: 1 VALU op per step ----
__device__ __forceinline__ float rowmax16(float v) {
    asm("v_max_num_f32_dpp %0, %0, %0 row_ror:8 row_mask:0xf bank_mask:0xf\n\t"
        "v_max_num_f32_dpp %0, %0, %0 row_ror:4 row_mask:0xf bank_mask:0xf\n\t"
        "v_max_num_f32_dpp %0, %0, %0 row_ror:2 row_mask:0xf bank_mask:0xf\n\t"
        "v_max_num_f32_dpp %0, %0, %0 row_ror:1 row_mask:0xf bank_mask:0xf"
        : "+v"(v));
    return v;
}
__device__ __forceinline__ float rowsum16(float v) {
    asm("v_add_f32_dpp %0, %0, %0 row_ror:8 row_mask:0xf bank_mask:0xf\n\t"
        "v_add_f32_dpp %0, %0, %0 row_ror:4 row_mask:0xf bank_mask:0xf\n\t"
        "v_add_f32_dpp %0, %0, %0 row_ror:2 row_mask:0xf bank_mask:0xf\n\t"
        "v_add_f32_dpp %0, %0, %0 row_ror:1 row_mask:0xf bank_mask:0xf"
        : "+v"(v));
    return v;
}

// ---------------- f32 -> bf16 conversion (vectorized) ----------------
__global__ __launch_bounds__(256) void cvt_f32_bf16(const float* __restrict__ s,
                                                    bf16_t* __restrict__ d, int n4)
{
    int i = blockIdx.x * 256 + threadIdx.x;
    if (i < n4) {
        float4 f = ((const float4*)s)[i];
        int j = i * 4;
        d[j+0] = (bf16_t)f.x; d[j+1] = (bf16_t)f.y;
        d[j+2] = (bf16_t)f.z; d[j+3] = (bf16_t)f.w;
    }
}

// ---------------- GEMM: out = oscale * (A[M,K] @ W[N,K]^T)  (K = N = DM) ----------------
// MODE 0: bf16 row-major [MM,DM]   (Q, K projections; Q gets oscale = 1/sqrt(Dh))
// MODE 1: bf16 transposed-per-head [B,H,Dh,T]  (V projection)
// MODE 2: f32 row-major [MM,DM]    (output projection)
template<int MODE>
__global__ __launch_bounds__(128) void gemm_xwt(const bf16_t* __restrict__ A,
                                                const bf16_t* __restrict__ W,
                                                void* __restrict__ outp,
                                                float oscale)
{
    const int lane = threadIdx.x & 31;
    const int wv   = threadIdx.x >> 5;
    const int wid  = blockIdx.x * 4 + wv;
    const int NB   = DM / 128;              // 8 n-blocks
    const int mTile = wid / NB;
    const int nBlk  = wid % NB;
    const int l15 = lane & 15;
    const int hl  = lane >> 4;              // half-wave id
    const int m   = mTile * 16 + l15;

    v8f acc[8] = {};

    const bf16_t* Arow = A + (size_t)m * DM;
    for (int k0 = 0; k0 < DM; k0 += 32) {
        // A fragment: lane<16 holds K 0..15, lanes 16-31 K 16..31 (contiguous, kb = hl*16)
        v16bf a;
        v8bf a0 = *(const v8bf*)(Arow + k0 + hl * 16);
        v8bf a1 = *(const v8bf*)(Arow + k0 + hl * 16 + 8);
#pragma unroll
        for (int i = 0; i < 8; ++i) { a[i] = a0[i]; a[8 + i] = a1[i]; }
#pragma unroll
        for (int t = 0; t < 8; ++t) {
            // B fragment: column n = W row n; 16 contiguous K, lanes 0-15 K 0..15, 16-31 K 16..31
            const bf16_t* wrow = W + (size_t)(nBlk * 128 + t * 16 + l15) * DM + k0 + hl * 16;
            v16bf b = *(const v16bf*)wrow;
            acc[t] = __builtin_amdgcn_wmma_f32_16x16x32_bf16(
                false, a, false, b, (short)0, acc[t], false, false);
        }
    }

#pragma unroll
    for (int t = 0; t < 8; ++t) {
        const int n = nBlk * 128 + t * 16 + l15;
#pragma unroll
        for (int r = 0; r < 8; ++r) {
            const int row = mTile * 16 + r + 8 * hl;   // C layout: VGPR r -> M=r (+8 for hi half)
            const float v = acc[t][r] * oscale;
            if constexpr (MODE == 0) {
                ((bf16_t*)outp)[(size_t)row * DM + n] = (bf16_t)v;
            } else if constexpr (MODE == 1) {
                const int h = n >> 6, dd = n & 63;
                const int b_ = row >> 11, t_ = row & (TT - 1);
                ((bf16_t*)outp)[(((size_t)(b_ * NH + h)) * DH + dd) * TT + t_] = (bf16_t)v;
            } else {
                ((float*)outp)[(size_t)row * DM + n] = v;
            }
        }
    }
}

// ---------------- Flash attention: one wave per (b, h, 16-query block) ----------------
// Q is pre-scaled by 1/sqrt(Dh) at projection time.
__global__ __launch_bounds__(32) void attn_kernel(const bf16_t* __restrict__ q,
                                                  const bf16_t* __restrict__ k,
                                                  const bf16_t* __restrict__ vt,
                                                  bf16_t* __restrict__ attOut)
{
    __shared__ bf16_t pl[16][34];   // P tile re-swizzle buffer (C-layout -> A-layout)

    const int lane = threadIdx.x & 31;
    int wid = blockIdx.x;                 // 0..B*H*(T/16)-1
    const int qb = wid % (TT / 16); wid /= (TT / 16);
    const int h  = wid % NH;
    const int b  = wid / NH;
    const int q0 = qb * 16;
    const int l15 = lane & 15;
    const int hl  = lane >> 4;

    // Q A-fragments (16 queries x 64 d), resident for whole key loop
    v16bf qa[2];
    const bf16_t* qrow = q + ((size_t)(b * TT + q0 + l15)) * DM + h * DH;
#pragma unroll
    for (int c = 0; c < 2; ++c) {
        v8bf a0 = *(const v8bf*)(qrow + c * 32 + hl * 16);
        v8bf a1 = *(const v8bf*)(qrow + c * 32 + hl * 16 + 8);
#pragma unroll
        for (int i = 0; i < 8; ++i) { qa[c][i] = a0[i]; qa[c][8 + i] = a1[i]; }
    }

    v8f acc[4] = {};                 // output accum: 16 queries x 64 d (4 C tiles)
    float mstate[8], lpart[8];       // lpart: per-lane PARTIAL row sums (reduced once at end)
#pragma unroll
    for (int r = 0; r < 8; ++r) { mstate[r] = -1e30f; lpart[r] = 0.f; }

    const int kbEnd = (q0 + 15) >> 5;                        // inclusive, 32 keys/iter
    const bf16_t* kbase = k  + ((size_t)b * TT) * DM + h * DH;
    const bf16_t* vbase = vt + (((size_t)(b * NH + h)) * DH) * TT;

    for (int kb = 0; kb <= kbEnd; ++kb) {
        const int kk0 = kb * 32;

        // S = Q @ K^T  (two 16-key tiles, Dh=64 -> 2 WMMA k-steps each)
        v8f s[2] = {};
#pragma unroll
        for (int t = 0; t < 2; ++t) {
            const bf16_t* kcol = kbase + ((size_t)(kk0 + t * 16 + l15)) * DM;
#pragma unroll
            for (int c = 0; c < 2; ++c) {
                v16bf bf = *(const v16bf*)(kcol + c * 32 + hl * 16);
                s[t] = __builtin_amdgcn_wmma_f32_16x16x32_bf16(
                    false, qa[c], false, bf, (short)0, s[t], false, false);
            }
        }

        // causal mask: only ever active on the diagonal block (wave-uniform branch)
        if (kb == kbEnd) {
#pragma unroll
            for (int t = 0; t < 2; ++t) {
                const int key = kk0 + t * 16 + l15;
#pragma unroll
                for (int r = 0; r < 8; ++r) {
                    const int row = q0 + r + 8 * hl;
                    if (key > row) s[t][r] = -1e30f;
                }
            }
        }

        // online softmax update; row-max via fused DPP VALU, row-sum deferred to end
#pragma unroll
        for (int r = 0; r < 8; ++r) {
            const float mx   = rowmax16(fmaxf(s[0][r], s[1][r]));
            const float mnew = fmaxf(mstate[r], mx);
            const float p0 = __expf(s[0][r] - mnew);
            const float p1 = __expf(s[1][r] - mnew);
            s[0][r] = p0; s[1][r] = p1;
            const float alpha = __expf(mstate[r] - mnew);
            lpart[r] = lpart[r] * alpha + (p0 + p1);
            mstate[r] = mnew;
#pragma unroll
            for (int t = 0; t < 4; ++t) acc[t][r] *= alpha;
        }

        // re-swizzle P (C-layout) into A-layout via LDS
        __syncthreads();
#pragma unroll
        for (int t = 0; t < 2; ++t)
#pragma unroll
            for (int r = 0; r < 8; ++r)
                pl[r + 8 * hl][t * 16 + l15] = (bf16_t)s[t][r];
        __syncthreads();

        v16bf pa;
        {
            const bf16_t* prow = &pl[l15][0];
#pragma unroll
            for (int i = 0; i < 8; ++i) {
                pa[i]     = prow[hl * 16 + i];
                pa[8 + i] = prow[hl * 16 + 8 + i];
            }
        }

        // acc += P @ V   (V stored [B,H,Dh,T] so B-fragment is contiguous along keys)
#pragma unroll
        for (int t = 0; t < 4; ++t) {
            const bf16_t* vcol = vbase + ((size_t)(t * 16 + l15)) * TT + kk0 + hl * 16;
            v16bf bv = *(const v16bf*)vcol;
            acc[t] = __builtin_amdgcn_wmma_f32_16x16x32_bf16(
                false, pa, false, bv, (short)0, acc[t], false, false);
        }
    }

    // reduce the deferred row sums once, normalize, store [B,T,H*Dh] bf16
    float linv[8];
#pragma unroll
    for (int r = 0; r < 8; ++r) linv[r] = 1.0f / rowsum16(lpart[r]);

#pragma unroll
    for (int t = 0; t < 4; ++t) {
        const int d = t * 16 + l15;
#pragma unroll
        for (int r = 0; r < 8; ++r) {
            const int row = q0 + r + 8 * hl;
            const float v = acc[t][r] * linv[r];
            attOut[((size_t)(b * TT + row)) * DM + h * DH + d] = (bf16_t)v;
        }
    }
}

// ---------------- launch ----------------
extern "C" void kernel_launch(void* const* d_in, const int* in_sizes, int n_in,
                              void* d_out, int out_size, void* d_ws, size_t ws_size,
                              hipStream_t stream) {
    (void)in_sizes; (void)n_in; (void)out_size; (void)ws_size;
    const float* x  = (const float*)d_in[0];
    const float* Wq = (const float*)d_in[1];
    const float* Wk = (const float*)d_in[2];
    const float* Wv = (const float*)d_in[3];
    const float* Wo = (const float*)d_in[4];

    char* ws = (char*)d_ws;
    bf16_t* xb  = (bf16_t*)(ws + 0);
    bf16_t* wqb = (bf16_t*)(ws + 8388608);
    bf16_t* wkb = (bf16_t*)(ws + 10485760);
    bf16_t* wvb = (bf16_t*)(ws + 12582912);
    bf16_t* wob = (bf16_t*)(ws + 14680064);
    bf16_t* qb  = (bf16_t*)(ws + 16777216);
    bf16_t* kb  = (bf16_t*)(ws + 25165824);
    bf16_t* vtb = (bf16_t*)(ws + 33554432);
    bf16_t* att = (bf16_t*)(ws + 41943040);

    // convert inputs to bf16
    cvt_f32_bf16<<<(MM * DM / 4 + 255) / 256, 256, 0, stream>>>(x, xb, MM * DM / 4);
    cvt_f32_bf16<<<(DM * DM / 4 + 255) / 256, 256, 0, stream>>>(Wq, wqb, DM * DM / 4);
    cvt_f32_bf16<<<(DM * DM / 4 + 255) / 256, 256, 0, stream>>>(Wk, wkb, DM * DM / 4);
    cvt_f32_bf16<<<(DM * DM / 4 + 255) / 256, 256, 0, stream>>>(Wv, wvb, DM * DM / 4);
    cvt_f32_bf16<<<(DM * DM / 4 + 255) / 256, 256, 0, stream>>>(Wo, wob, DM * DM / 4);

    // projections: 2048 waves = 512 blocks x 4 waves
    const int gemmBlocks = (MM / 16) * (DM / 128) / 4;
    const float qscale = 0.125f;   // 1/sqrt(Dh)
    gemm_xwt<0><<<gemmBlocks, 128, 0, stream>>>(xb, wqb, qb, qscale);
    gemm_xwt<0><<<gemmBlocks, 128, 0, stream>>>(xb, wkb, kb, 1.0f);
    gemm_xwt<1><<<gemmBlocks, 128, 0, stream>>>(xb, wvb, vtb, 1.0f);

    // flash attention: one wave per (b,h,16-query block)
    attn_kernel<<<BB * NH * (TT / 16), 32, 0, stream>>>(qb, kb, vtb, att);

    // output projection -> f32
    gemm_xwt<2><<<gemmBlocks, 128, 0, stream>>>(att, wob, (float*)d_out, 1.0f);
}
